// AttentionBlock_24352464569572
// MI455X (gfx1250) — matrix operand order repaired and checked by reading, not compiled, not run on hardware
//
#include <hip/hip_runtime.h>
#include <hip/hip_bf16.h>

typedef __attribute__((ext_vector_type(16))) _Float16 v16h;
typedef __attribute__((ext_vector_type(8)))  _Float16 v8h;
typedef __attribute__((ext_vector_type(8)))  float    v8f;

#define B_     4
#define C_     256
#define HEADS_ 4
#define D_     64
#define N_     4096
#define G_     8
#define CPG_   32
#define EPS_   1e-5f

// ---------------- CDNA5 async global->LDS DMA (ASYNCcnt-tracked) ------------

__device__ __forceinline__ uint32_t lds_addr(const void* p) {
  return (uint32_t)(uintptr_t)p;     // low 32 bits of generic ptr = LDS byte addr
}

__device__ __forceinline__ void async_b128(uint32_t lds_byte, const void* g) {
  asm volatile("global_load_async_to_lds_b128 %0, %1, off"
               :: "v"(lds_byte), "v"(g) : "memory");
}

template <int NMAX>
__device__ __forceinline__ void wait_async() {
  asm volatile("s_wait_asynccnt %0" :: "i"(NMAX) : "memory");
}

// ---------------- WMMA fragment helpers (CDNA5 documented layouts) ----------

__device__ __forceinline__ v16h frag_a_rm(const _Float16* a, int lda) {
  const int lane = threadIdx.x & 31;
  const int r = lane & 15, hh = lane >> 4;
  v16h f;
#pragma unroll
  for (int e = 0; e < 16; ++e) {
    const int k = ((e >> 3) << 4) + (hh << 3) + (e & 7);
    f[e] = a[r * lda + k];
  }
  return f;
}

// B fragment (32x16 KxN) from column-major storage bt[n][k] (k contiguous).
__device__ __forceinline__ v16h frag_b_cm(const _Float16* bt, int ldbt) {
  const int lane = threadIdx.x & 31;
  const int n = lane & 15, hh = lane >> 4;
  v16h f;
#pragma unroll
  for (int e = 0; e < 16; ++e) f[e] = bt[n * ldbt + ((e >> 3) << 4) + (hh << 3) + (e & 7)];
  return f;
}

__device__ __forceinline__ v8f wmma16(v16h a, v16h b, v8f c) {
  return __builtin_amdgcn_wmma_f32_16x16x32_f16(false, a, false, b,
                                                (short)0, c, false, false);
}

// ---------------- weight f32 -> f16 pre-convert (wq|wk|wv|wp stacked) -------

__global__ void __launch_bounds__(256) wcvt_kernel(const float* __restrict__ a,
                                                   const float* __restrict__ b,
                                                   const float* __restrict__ c,
                                                   const float* __restrict__ d,
                                                   _Float16* __restrict__ o) {
  const int i = blockIdx.x * 256 + threadIdx.x;
  o[i]          = (_Float16)a[i];
  o[i + 65536]  = (_Float16)b[i];
  o[i + 131072] = (_Float16)c[i];
  o[i + 196608] = (_Float16)d[i];
}

// ---------------- GroupNorm: (b,256,4096) f32 -> f16 ------------------------

__global__ void __launch_bounds__(256) gn_kernel(const float* __restrict__ x,
                                                 const float* __restrict__ gw,
                                                 const float* __restrict__ gb,
                                                 _Float16* __restrict__ xn) {
  const int g = blockIdx.x, b = blockIdx.y, tid = threadIdx.x;
  const size_t base = ((size_t)b * C_ + g * CPG_) * N_;
  const int cnt = CPG_ * N_;
  float s = 0.f, sq = 0.f;
  for (int i = tid; i < cnt; i += 256) {
    const float v = x[base + i];
    s += v; sq += v * v;
  }
  __shared__ float rs[256], rq[256];
  rs[tid] = s; rq[tid] = sq;
  __syncthreads();
  for (int st = 128; st > 0; st >>= 1) {
    if (tid < st) { rs[tid] += rs[tid + st]; rq[tid] += rq[tid + st]; }
    __syncthreads();
  }
  const float mean = rs[0] / (float)cnt;
  const float var  = rq[0] / (float)cnt - mean * mean;
  const float inv  = rsqrtf(var + EPS_);
  for (int i = tid; i < cnt; i += 256) {
    const int c = g * CPG_ + (i >> 12);
    xn[base + i] = (_Float16)((x[base + i] - mean) * inv * gw[c] + gb[c]);
  }
}

// ------- GEMM: out_s = W_s(256x256) @ X(b,256,4096), s<S --------------------
// !PROJ (S=3): X = xn [c][n]; writes qT/kT transposed [n][d] (q scaled 1/8),
//              v row-major [c][n].
// PROJ  (S=1): X = abT [n][c] (async-staged); +bias +residual, f32 out.

template <bool PROJ>
__global__ void __launch_bounds__(256) gemm256_kernel(
    const _Float16* __restrict__ Wh,
    const float* __restrict__ b0, const float* __restrict__ b1,
    const float* __restrict__ b2,
    const _Float16* __restrict__ X,
    _Float16* __restrict__ qT, _Float16* __restrict__ kT,
    _Float16* __restrict__ vO,
    const float* __restrict__ resid, float* __restrict__ outF) {
  constexpr int S = PROJ ? 1 : 3;
  const int ntb = blockIdx.x, mtb = blockIdx.y, b = blockIdx.z;
  const int tid = threadIdx.x, w = tid >> 5, lane = tid & 31;
  const int nlane = lane & 15, hh = lane >> 4;

  __shared__ _Float16 As[S][128][40];
  __shared__ _Float16 Bst[128][40];          // X tile column-major [n][k]

  const int m0 = mtb * 128, n0 = ntb * 128;
  const size_t xoff = (size_t)b * C_ * N_;
  const uint32_t as_base  = lds_addr(&As[0][0][0]);
  const uint32_t bst_base = lds_addr(&Bst[0][0]);

  v8f acc[S][8];
#pragma unroll
  for (int s = 0; s < S; ++s)
#pragma unroll
    for (int nf = 0; nf < 8; ++nf)
#pragma unroll
      for (int i = 0; i < 8; ++i) acc[s][nf][i] = 0.f;

  for (int kt = 0; kt < C_ / 32; ++kt) {
    // weight tiles: async DMA, 16B chunks
#pragma unroll
    for (int it = 0; it < S * 2; ++it) {
      const int c  = tid + 256 * it;
      const int s  = c >> 9, rm = (c >> 2) & 127, ch = c & 3;
      async_b128(as_base + (uint32_t)(((s * 128 + rm) * 40 + ch * 8) * 2),
                 Wh + (size_t)s * 65536 + (size_t)(m0 + rm) * C_ +
                     kt * 32 + ch * 8);
    }
    if constexpr (PROJ) {
      // X is [n][c]: async DMA straight into the column-major tile
#pragma unroll
      for (int it = 0; it < 2; ++it) {
        const int c = tid + 256 * it;
        const int n = c >> 2, ch = c & 3;
        async_b128(bst_base + (uint32_t)((n * 40 + ch * 8) * 2),
                   X + xoff + (size_t)(n0 + n) * C_ + kt * 32 + ch * 8);
      }
    } else {
      // X is [k][n]: transpose-stage (coalesced global reads along n)
#pragma unroll
      for (int j = 0; j < 16; ++j) {
        const int i = tid + 256 * j;
        const int n = i & 127, k = i >> 7;
        Bst[n][k] = X[xoff + (size_t)(kt * 32 + k) * N_ + n0 + n];
      }
    }
    wait_async<0>();
    __syncthreads();

    v16h a[S];
#pragma unroll
    for (int s = 0; s < S; ++s) a[s] = frag_a_rm(&As[s][w * 16][0], 40);
#pragma unroll
    for (int nf = 0; nf < 8; ++nf) {
      v16h bf = frag_b_cm(&Bst[nf * 16][0], 40);
#pragma unroll
      for (int s = 0; s < S; ++s) acc[s][nf] = wmma16(a[s], bf, acc[s][nf]);
    }
    __syncthreads();
  }

  const int d0   = (m0 + w * 16 + hh * 8) & 63;   // head-dim base (contig in i)
  const int head = (m0 + w * 16 + hh * 8) >> 6;
  const float* biases[3] = {b0, b1, b2};
#pragma unroll
  for (int s = 0; s < S; ++s) {
#pragma unroll
    for (int nf = 0; nf < 8; ++nf) {
      const int ncol = n0 + nf * 16 + nlane;
      if (PROJ) {
#pragma unroll
        for (int i = 0; i < 8; ++i) {
          const int mrow = m0 + w * 16 + hh * 8 + i;
          const size_t off = xoff + (size_t)mrow * N_ + ncol;
          outF[off] = acc[s][nf][i] + biases[s][mrow] + resid[off];
        }
      } else if (s < 2) {
        // q (scaled by 1/sqrt(d)) and k -> transposed [n][d], one 16B store
        const float scl = (s == 0) ? 0.125f : 1.0f;
        v8h pk;
#pragma unroll
        for (int i = 0; i < 8; ++i) {
          const int mrow = m0 + w * 16 + hh * 8 + i;
          pk[i] = (_Float16)((acc[s][nf][i] + biases[s][mrow]) * scl);
        }
        _Float16* dst = (s == 0 ? qT : kT) +
            ((size_t)(b * HEADS_ + head) * N_ + ncol) * 64 + d0;
        *(v8h*)dst = pk;
      } else {
        // v stays row-major [c][n]
#pragma unroll
        for (int i = 0; i < 8; ++i) {
          const int mrow = m0 + w * 16 + hh * 8 + i;
          vO[xoff + (size_t)mrow * N_ + ncol] =
              (_Float16)(acc[s][nf][i] + biases[s][mrow]);
        }
      }
    }
  }
}

// ------------- Flash attention: per (b, head, 64-query tile) ----------------
// qT/kT: [bh][n][d]; v: [c][n]; output oT: [b][n][c] (feeds proj GEMM).

__global__ void __launch_bounds__(256) attn_kernel(
    const _Float16* __restrict__ qT, const _Float16* __restrict__ kT,
    const _Float16* __restrict__ vb, _Float16* __restrict__ oT) {
  const int qt = blockIdx.x, head = blockIdx.y, b = blockIdx.z;
  const int tid = threadIdx.x, w = tid >> 5, lane = tid & 31;
  const int nlane = lane & 15, hh = lane >> 4;

  __shared__ _Float16 Qt[64][72];        // [q][d]
  __shared__ _Float16 Kt[2][64][72];     // [key][d], double-buffered
  __shared__ _Float16 KV[64][72];        // V [d][key]
  __shared__ float    Ssc[64][64];
  __shared__ _Float16 Pm[64][64];
  __shared__ float    mrun[64], lrun[64], fac[64];

  const size_t bh = (size_t)(b * HEADS_ + head);
  const _Float16* qg = qT + bh * N_ * 64;
  const _Float16* kg = kT + bh * N_ * 64;
  const _Float16* vg = vb + ((size_t)b * C_ + head * D_) * N_;
  const int q0 = qt * 64;
  const uint32_t qt_base = lds_addr(&Qt[0][0]);
  const uint32_t kt_b[2] = {lds_addr(&Kt[0][0][0]), lds_addr(&Kt[1][0][0])};
  const uint32_t kv_base = lds_addr(&KV[0][0]);

#pragma unroll
  for (int it = 0; it < 2; ++it) {       // Q tile (contiguous [q][d])
    const int c = tid + 256 * it;
    const int r = c >> 3, ch = c & 7;
    async_b128(qt_base + (uint32_t)((r * 72 + ch * 8) * 2),
               qg + (size_t)(q0 + r) * 64 + ch * 8);
  }
#pragma unroll
  for (int it = 0; it < 2; ++it) {       // K tile 0
    const int c = tid + 256 * it;
    const int r = c >> 3, ch = c & 7;
    async_b128(kt_b[0] + (uint32_t)((r * 72 + ch * 8) * 2),
               kg + (size_t)r * 64 + ch * 8);
  }
  if (tid < 64) { mrun[tid] = -3.0e38f; lrun[tid] = 0.f; }
  wait_async<2>();                       // Q landed (K0 may be in flight)
  __syncthreads();

  const int mt = w & 3, wdh = w >> 2;
  v16h aq[2];                            // loop-invariant Q fragments
  aq[0] = frag_a_rm(&Qt[mt * 16][0], 72);
  aq[1] = frag_a_rm(&Qt[mt * 16][32], 72);
  v8f Of[2];
#pragma unroll
  for (int j = 0; j < 2; ++j)
#pragma unroll
    for (int i = 0; i < 8; ++i) Of[j][i] = 0.f;

  for (int ktile = 0; ktile < N_ / 64; ++ktile) {
    const int k0 = ktile * 64;
    const int cur = ktile & 1, nxt = cur ^ 1;
    const int kn0 = ((ktile + 1) & 63) * 64;   // wrap: last iter reloads tile 0
    // V(i) + K(i+1): fully overlapped with the math below
#pragma unroll
    for (int it = 0; it < 2; ++it) {
      const int c = tid + 256 * it;
      const int r = c >> 3, ch = c & 7;
      async_b128(kv_base + (uint32_t)((r * 72 + ch * 8) * 2),
                 vg + (size_t)r * N_ + k0 + ch * 8);
    }
#pragma unroll
    for (int it = 0; it < 2; ++it) {
      const int c = tid + 256 * it;
      const int r = c >> 3, ch = c & 7;
      async_b128(kt_b[nxt] + (uint32_t)((r * 72 + ch * 8) * 2),
                 kg + (size_t)(kn0 + r) * 64 + ch * 8);
    }
    wait_async<4>();                     // in-order: K(i) retired
    __syncthreads();

    // S = Q^T K (q pre-scaled by 1/8)
#pragma unroll
    for (int j = 0; j < 2; ++j) {
      const int nt = wdh * 2 + j;
      v8f sacc;
#pragma unroll
      for (int i = 0; i < 8; ++i) sacc[i] = 0.f;
      sacc = wmma16(aq[0], frag_b_cm(&Kt[cur][nt * 16][0], 72), sacc);
      sacc = wmma16(aq[1], frag_b_cm(&Kt[cur][nt * 16][32], 72), sacc);
#pragma unroll
      for (int i = 0; i < 8; ++i)
        Ssc[mt * 16 + hh * 8 + i][nt * 16 + nlane] = sacc[i];
    }
    __syncthreads();

    // online softmax: 4 threads/row, vector LDS accesses, shfl reductions
    {
      const int r = tid >> 2, qd = tid & 3;
      float sv[16];
#pragma unroll
      for (int t = 0; t < 4; ++t)
        *(float4*)&sv[t * 4] = *(const float4*)&Ssc[r][qd * 16 + t * 4];
      const float mold = mrun[r];
      float mx = mold;
#pragma unroll
      for (int t = 0; t < 16; ++t) mx = fmaxf(mx, sv[t]);
      mx = fmaxf(mx, __shfl_xor(mx, 1, 32));
      mx = fmaxf(mx, __shfl_xor(mx, 2, 32));
      float sum = 0.f;
      v8h pk0, pk1;
#pragma unroll
      for (int t = 0; t < 8; ++t) {
        const float p = __expf(sv[t] - mx);
        pk0[t] = (_Float16)p; sum += p;
      }
#pragma unroll
      for (int t = 0; t < 8; ++t) {
        const float p = __expf(sv[8 + t] - mx);
        pk1[t] = (_Float16)p; sum += p;
      }
      *(v8h*)&Pm[r][qd * 16]     = pk0;
      *(v8h*)&Pm[r][qd * 16 + 8] = pk1;
      sum += __shfl_xor(sum, 1, 32);
      sum += __shfl_xor(sum, 2, 32);
      if (qd == 0) {
        const float f = __expf(mold - mx);
        mrun[r] = mx; lrun[r] = lrun[r] * f + sum; fac[r] = f;
      }
    }
    wait_async<2>();                     // V(i) retired (K(i+1) in flight)
    __syncthreads();

    // O = O*fac + P V^T
    {
      float fr[8];
      *(float4*)&fr[0] = *(const float4*)&fac[mt * 16 + hh * 8];
      *(float4*)&fr[4] = *(const float4*)&fac[mt * 16 + hh * 8 + 4];
#pragma unroll
      for (int j = 0; j < 2; ++j)
#pragma unroll
        for (int i = 0; i < 8; ++i) Of[j][i] *= fr[i];
    }
#pragma unroll
    for (int kbc = 0; kbc < 64; kbc += 32) {
      v16h a = frag_a_rm(&Pm[mt * 16][kbc], 64);
#pragma unroll
      for (int j = 0; j < 2; ++j) {
        const int dt = wdh * 2 + j;
        Of[j] = wmma16(a, frag_b_cm(&KV[dt * 16][kbc], 72), Of[j]);
      }
    }
    __syncthreads();
  }

  // finalize: O/l -> LDS bounce -> transposed coalesced store [n][c]
  {
    float lr[8];
    *(float4*)&lr[0] = *(const float4*)&lrun[mt * 16 + hh * 8];
    *(float4*)&lr[4] = *(const float4*)&lrun[mt * 16 + hh * 8 + 4];
#pragma unroll
    for (int j = 0; j < 2; ++j) {
      const int dcol = (wdh * 2 + j) * 16 + nlane;
#pragma unroll
      for (int i = 0; i < 8; ++i)
        Pm[mt * 16 + hh * 8 + i][dcol] = (_Float16)(Of[j][i] / lr[i]);
    }
  }
  __syncthreads();
#pragma unroll
  for (int j = 0; j < 16; ++j) {
    const int i = tid + 256 * j;
    const int d = i & 63, q = i >> 6;
    oT[((size_t)b * N_ + q0 + q) * C_ + head * D_ + d] = Pm[q][d];
  }
}

// ---------------------------------------------------------------------------

extern "C" void kernel_launch(void* const* d_in, const int* in_sizes, int n_in,
                              void* d_out, int out_size, void* d_ws, size_t ws_size,
                              hipStream_t stream) {
  (void)in_sizes; (void)n_in; (void)out_size; (void)ws_size;
  const float* x  = (const float*)d_in[0];
  const float* gw = (const float*)d_in[1];
  const float* gb = (const float*)d_in[2];
  const float* wq = (const float*)d_in[3];
  const float* bq = (const float*)d_in[4];
  const float* wk = (const float*)d_in[5];
  const float* bk = (const float*)d_in[6];
  const float* wv = (const float*)d_in[7];
  const float* bv = (const float*)d_in[8];
  const float* wp = (const float*)d_in[9];
  const float* bp = (const float*)d_in[10];
  float* out = (float*)d_out;

  const size_t elems = (size_t)B_ * C_ * N_;
  _Float16* xn  = (_Float16*)d_ws;
  _Float16* qTb = xn  + elems;    // [bh][n][d] transposed
  _Float16* kTb = qTb + elems;    // [bh][n][d] transposed
  _Float16* vbf = kTb + elems;    // [c][n]
  _Float16* abT = vbf + elems;    // [n][c] transposed attention output
  _Float16* wh  = abT + elems;    // 4 x 65536 f16 weights

  const dim3 blk(256);
  wcvt_kernel<<<dim3(256), blk, 0, stream>>>(wq, wk, wv, wp, wh);
  gn_kernel<<<dim3(G_, B_), blk, 0, stream>>>(x, gw, gb, xn);

  const dim3 ggrid(N_ / 128, C_ / 128, B_);
  gemm256_kernel<false><<<ggrid, blk, 0, stream>>>(
      wh, bq, bk, bv, xn, qTb, kTb, vbf, nullptr, nullptr);

  attn_kernel<<<dim3(N_ / 64, HEADS_, B_), blk, 0, stream>>>(qTb, kTb, vbf, abT);

  gemm256_kernel<true><<<ggrid, blk, 0, stream>>>(
      wh + 3 * 65536, bp, nullptr, nullptr, abT, nullptr, nullptr, nullptr,
      x, out);
}
